// LinearAttention_34187939676583
// MI455X (gfx1250) — matrix operand order repaired and checked by reading, not compiled, not run on hardware
//
#include <hip/hip_runtime.h>
#include <cstdint>
#include <cstddef>

// ---------------------------------------------------------------------------
// MI455X (gfx1250) linear-attention pipeline, bf16 WMMA + f32 softmax/norms.
// Round 2: native bf16 converts, async-to-LDS staging, global prefetch.
// ---------------------------------------------------------------------------

typedef __attribute__((ext_vector_type(16))) __bf16 v16bf;
typedef __attribute__((ext_vector_type(8)))  float  v8f;

union V16 { v16bf v; unsigned short s[16]; };

#define NB   16          // batch
#define NC   256         // channels (DIM)
#define NN   16384       // tokens per image (128*128)
#define NH   4           // heads
#define NDH  32          // dim_head
#define NHID 128         // heads*dim_head
#define NO3  768         // 3*DIM
#define NMEM 4           // memory kv tokens

__device__ __forceinline__ unsigned short f2bfu(float f) {
  union { __bf16 b; unsigned short s; } u;
  u.b = (__bf16)f;                        // hardware cvt on gfx1250
  return u.s;
}
__device__ __forceinline__ float bfu2f(unsigned short h) {
  union { __bf16 b; unsigned short s; } u;
  u.s = h;
  return (float)u.b;
}

__device__ __forceinline__ void wait_asynccnt0() {
#if __has_builtin(__builtin_amdgcn_s_wait_asynccnt)
  __builtin_amdgcn_s_wait_asynccnt(0);
#else
  asm volatile("s_wait_asynccnt 0" ::: "memory");
#endif
}

// async byte-copy of one dword global -> LDS (ASYNCcnt-tracked)
__device__ __forceinline__ void async_g2l_b32(unsigned lds_addr, const void* g) {
  asm volatile("global_load_async_to_lds_b32 %0, %1, off"
               :: "v"(lds_addr), "v"((unsigned long long)(uintptr_t)g)
               : "memory");
}

// A-fragment: 16x32 bf16, row-major source [M][K] with leading dim ld.
// lane: M = m0 + (lane&15); K runs: kh..kh+7 and 16+kh..16+kh+7, kh=(lane>>4)*8
__device__ __forceinline__ v16bf frag_a(const unsigned short* base, int ld,
                                        int m0, int k0, int lane) {
  const unsigned short* r = base + (size_t)(m0 + (lane & 15)) * ld + k0;
  const int kh = (lane >> 4) * 8;
  V16 t;
#pragma unroll
  for (int j = 0; j < 8; ++j) t.s[j] = r[kh + j];
#pragma unroll
  for (int j = 0; j < 8; ++j) t.s[8 + j] = r[16 + kh + j];
  return t.v;
}

// B-fragment: 32x16 bf16 (KxN), source stored N-major: [N][K] with leading dim ld.
// lane: N = n0 + (lane&15); K runs: kh..kh+7 and 16+kh..16+kh+7, kh=(lane>>4)*8
__device__ __forceinline__ v16bf frag_bt(const unsigned short* base, int ld,
                                         int n0, int k0, int lane) {
  const unsigned short* r =
      base + (size_t)(n0 + (lane & 15)) * ld + k0;
  const int kh = (lane >> 4) * 8;
  V16 t;
#pragma unroll
  for (int j = 0; j < 8; ++j) t.s[j] = r[kh + j];
#pragma unroll
  for (int j = 0; j < 8; ++j) t.s[8 + j] = r[16 + kh + j];
  return t.v;
}

__device__ __forceinline__ v8f wmma_bf16(v16bf a, v16bf b, v8f c) {
  return __builtin_amdgcn_wmma_f32_16x16x32_bf16(false, a, false, b,
                                                 (short)0, c, false, false);
}

// ---------------------------------------------------------------------------
// Kernel 1: weight prep. Fold (gamma1+1)*sqrt(DIM) into w_qkv columns, bf16.
// ---------------------------------------------------------------------------
__global__ __launch_bounds__(256) void prep_kernel(
    const float* __restrict__ w_qkv, const float* __restrict__ g1,
    const float* __restrict__ w_out,
    unsigned short* __restrict__ wqkv_bf, unsigned short* __restrict__ wout_bf) {
  int i = blockIdx.x * 256 + threadIdx.x;
  if (i < NO3 * NC) {
    int c = i & (NC - 1);
    wqkv_bf[i] = f2bfu(w_qkv[i] * (g1[c] + 1.0f) * 16.0f);  // sqrt(256)=16
  }
  int j = i - NO3 * NC;
  if (j >= 0 && j < NC * NHID) wout_bf[j] = f2bfu(w_out[j]);
}

// ---------------------------------------------------------------------------
// Kernel 2: qkv GEMM. Per block: one batch, 32-token tile, M=768, K=256.
// x staged f32->bf16 transposed into LDS; per-token sum(x^2) fused in; the
// per-pixel inverse RMS norm is applied in the epilogue (factors out of the
// channel contraction). Outputs q,k,v as bf16 [b][h*32+d][n].
// ---------------------------------------------------------------------------
__global__ __launch_bounds__(256) void qkv_kernel(
    const float* __restrict__ x, const unsigned short* __restrict__ wqkv_bf,
    unsigned short* __restrict__ qw, unsigned short* __restrict__ kw,
    unsigned short* __restrict__ vw) {
  __shared__ unsigned short xT[32][40];   // [n][c] bf16 tile, padded pitch
  __shared__ float sqp[8][32];
  __shared__ float invs[32];

  const int b   = blockIdx.y;
  const int n0  = blockIdx.x * 32;
  const int wv  = threadIdx.x >> 5;
  const int lane = threadIdx.x & 31;
  const int tn  = threadIdx.x & 31;       // token this thread owns for sumsq
  const int tg  = threadIdx.x >> 5;

  v8f acc[6][2] = {};
  float sq = 0.0f;

#pragma unroll 1
  for (int kc = 0; kc < 8; ++kc) {
    const int c0 = kc * 32;
    // stage 32x32 f32 chunk -> bf16 transposed, accumulate sum of squares
#pragma unroll
    for (int r = 0; r < 4; ++r) {
      int i = threadIdx.x + 256 * r;
      int c = i >> 5;             // 0..31
      int n = i & 31;             // == tn for every r
      float xv = x[((size_t)b * NC + c0 + c) * NN + n0 + n];
      xT[n][c] = f2bfu(xv);
      sq += xv * xv;
    }
    __syncthreads();

    v16bf b0 = frag_bt(&xT[0][0], 40, 0, 0, lane);
    v16bf b1 = frag_bt(&xT[0][0], 40, 16, 0, lane);
#pragma unroll
    for (int ms = 0; ms < 6; ++ms) {
      int m0 = (wv * 6 + ms) * 16;
      v16bf a = frag_a(wqkv_bf, NC, m0, c0, lane);
      acc[ms][0] = wmma_bf16(a, b0, acc[ms][0]);
      acc[ms][1] = wmma_bf16(a, b1, acc[ms][1]);
    }
    __syncthreads();
  }

  // reduce per-token sum of squares -> 1/max(||x||, eps)
  sqp[tg][tn] = sq;
  __syncthreads();
  if (threadIdx.x < 32) {
    float s = 0.0f;
#pragma unroll
    for (int g = 0; g < 8; ++g) s += sqp[g][threadIdx.x];
    invs[threadIdx.x] = 1.0f / fmaxf(sqrtf(s), 1e-12f);
  }
  __syncthreads();

  // epilogue: scale by per-token inv norm, scatter to q/k/v bf16
#pragma unroll
  for (int ms = 0; ms < 6; ++ms) {
#pragma unroll
    for (int ns = 0; ns < 2; ++ns) {
      int nl = ns * 16 + (lane & 15);
      float inv = invs[nl];
      size_t ng = (size_t)n0 + nl;
#pragma unroll
      for (int r = 0; r < 8; ++r) {
        int o = (wv * 6 + ms) * 16 + r + 8 * (lane >> 4);
        float v = acc[ms][ns][r] * inv;
        unsigned short* dst = (o < 256) ? qw : ((o < 512) ? kw : vw);
        dst[((size_t)b * NHID + (o & 127)) * NN + ng] = f2bfu(v);
      }
    }
  }
}

// ---------------------------------------------------------------------------
// Kernel 3: per-(b,h,d) row max of k over all tokens + memory tokens.
// Vectorized uint4 loads: 8 bf16 per lane per iteration.
// ---------------------------------------------------------------------------
__global__ __launch_bounds__(256) void rowmax_kernel(
    const unsigned short* __restrict__ kw, const float* __restrict__ memkv,
    float* __restrict__ rowmax) {
  int row = blockIdx.x * 8 + (threadIdx.x >> 5);   // b*128 + h*32 + d
  int lane = threadIdx.x & 31;
  int b = row >> 7;
  int hd = row & 127;
  const uint4* kr4 = (const uint4*)(kw + ((size_t)b * NHID + hd) * NN);
  float m = -3.0e38f;
  for (int i = lane; i < NN / 8; i += 32) {
    uint4 p = kr4[i];
    unsigned w0 = p.x, w1 = p.y, w2 = p.z, w3 = p.w;
    m = fmaxf(m, bfu2f((unsigned short)(w0 & 0xFFFF)));
    m = fmaxf(m, bfu2f((unsigned short)(w0 >> 16)));
    m = fmaxf(m, bfu2f((unsigned short)(w1 & 0xFFFF)));
    m = fmaxf(m, bfu2f((unsigned short)(w1 >> 16)));
    m = fmaxf(m, bfu2f((unsigned short)(w2 & 0xFFFF)));
    m = fmaxf(m, bfu2f((unsigned short)(w2 >> 16)));
    m = fmaxf(m, bfu2f((unsigned short)(w3 & 0xFFFF)));
    m = fmaxf(m, bfu2f((unsigned short)(w3 >> 16)));
  }
#pragma unroll
  for (int off = 16; off; off >>= 1) m = fmaxf(m, __shfl_xor(m, off, 32));
  if (lane == 0) {
#pragma unroll
    for (int j = 0; j < NMEM; ++j) m = fmaxf(m, memkv[hd * NMEM + j]);
    rowmax[row] = m;
  }
}

// ---------------------------------------------------------------------------
// Kernel 4: context[b][h] = softmax_n(k) @ v^T  (32x32), via WMMA over token
// chunks. A = exp(k-m) built in VALU (bf16 frag); Z accumulated lane-locally
// (each A lane owns one d-row). Memory kv tokens folded in epilogue; result
// stored pre-divided by Z.
// ---------------------------------------------------------------------------
__global__ __launch_bounds__(128) void context_kernel(
    const unsigned short* __restrict__ kw, const unsigned short* __restrict__ vw,
    const float* __restrict__ memkv, const float* __restrict__ rowmax,
    float* __restrict__ ctx) {
  __shared__ float zl[32];
  const int bh = blockIdx.x;
  const int b = bh >> 2, h = bh & 3;
  const int wv = threadIdx.x >> 5, lane = threadIdx.x & 31;
  const int msub = wv >> 1, esub = wv & 1;

  if (threadIdx.x < 32) zl[threadIdx.x] = 0.0f;
  __syncthreads();

  const unsigned short* kb_ = kw + ((size_t)b * NHID + h * NDH) * NN;
  const unsigned short* vb_ = vw + ((size_t)b * NHID + h * NDH) * NN;
  const int d_lane = msub * 16 + (lane & 15);
  const float md = rowmax[bh * NDH + d_lane];
  const int kh = (lane >> 4) * 8;
  const unsigned short* krow = kb_ + (size_t)d_lane * NN;
  const unsigned short* vrow =
      vb_ + (size_t)(esub * 16 + (lane & 15)) * NN + (lane >> 4) * 16;

  v8f acc = {};
  float z = 0.0f;
#pragma unroll 1
  for (int nc = 0; nc < NN; nc += 32) {
    if (nc + 32 < NN) {                       // stream-ahead hints
      __builtin_prefetch(krow + nc + 32, 0, 3);
      __builtin_prefetch(vrow + nc + 32, 0, 3);
    }
    V16 a;
#pragma unroll
    for (int j = 0; j < 8; ++j) {
      float t = __expf(bfu2f(krow[nc + kh + j]) - md);
      z += t; a.s[j] = f2bfu(t);
    }
#pragma unroll
    for (int j = 0; j < 8; ++j) {
      float t = __expf(bfu2f(krow[nc + 16 + kh + j]) - md);
      z += t; a.s[8 + j] = f2bfu(t);
    }
    v16bf bf = frag_bt(vb_, NN, esub * 16, nc, lane);
    acc = wmma_bf16(a.v, bf, acc);
  }

  if (esub == 0) atomicAdd(&zl[d_lane], z);
  __syncthreads();
  if (threadIdx.x < 32) {          // fold memory tokens into Z
    int d = threadIdx.x;
    float mdd = rowmax[bh * NDH + d];
    float zz = zl[d];
#pragma unroll
    for (int j = 0; j < NMEM; ++j)
      zz += __expf(memkv[(h * NDH + d) * NMEM + j] - mdd);
    zl[d] = zz;
  }
  __syncthreads();

  const int e_lane = esub * 16 + (lane & 15);
#pragma unroll
  for (int r = 0; r < 8; ++r) {
    int d = msub * 16 + r + 8 * (lane >> 4);
    float mdd = rowmax[bh * NDH + d];
    float v = acc[r];
#pragma unroll
    for (int j = 0; j < NMEM; ++j)
      v += __expf(memkv[(h * NDH + d) * NMEM + j] - mdd) *
           memkv[NH * NDH * NMEM + (h * NDH + e_lane) * NMEM + j];
    ctx[((size_t)bh * NDH + d) * NDH + e_lane] = v / zl[d];
  }
}

// ---------------------------------------------------------------------------
// Kernel 5: per 16-token tile: async-copy q tile to LDS (ASYNCcnt path),
// q-softmax over d, out = ctx^T @ q_soft (WMMA), y = W_out @ out + b_out
// (WMMA, K=128), per-pixel RMS norm, fp32 store.
// ---------------------------------------------------------------------------
__global__ __launch_bounds__(256) void attn_out_kernel(
    const unsigned short* __restrict__ qw, const float* __restrict__ ctx,
    const unsigned short* __restrict__ wout_bf, const float* __restrict__ bout,
    const float* __restrict__ g2, float* __restrict__ out) {
  __shared__ unsigned short q_raw[NH][NDH][16];  // [h][d][n] raw bf16 (4 KiB)
  __shared__ unsigned short ctxT[NH][NDH][NDH];  // [h][e][d] bf16
  __shared__ unsigned short qs[NH][16][NDH];     // softmaxed, bf16 [h][n][d]
  __shared__ unsigned short onb[16][132];        // [n][e] bf16
  __shared__ float yl[NC][17];                   // [o][n]
  __shared__ float part[16][16];
  __shared__ float invn[16];

  const int b = blockIdx.y;
  const int n0 = blockIdx.x * 16;
  const int wv = threadIdx.x >> 5, lane = threadIdx.x & 31;

  // async copy q tile (1024 dwords) global -> LDS, raw layout
  {
    unsigned lds_base = (unsigned)(uintptr_t)(&q_raw[0][0][0]);
    const char* gbase = (const char*)(qw + ((size_t)b * NHID) * NN + n0);
#pragma unroll
    for (int r = 0; r < 4; ++r) {
      int idx = threadIdx.x + 256 * r;   // dword index 0..1023
      int row = idx >> 3;                // (h*32+d) row, 8 dwords each
      int dw  = idx & 7;
      async_g2l_b32(lds_base + row * 32 + dw * 4,
                    gbase + (size_t)row * (NN * 2) + dw * 4);
    }
  }

  // stage context transposed as bf16 (regular loads, overlap with async)
  for (int i = threadIdx.x; i < NH * NDH * NDH; i += 256) {
    int h = i >> 10, d = (i >> 5) & 31, e = i & 31;
    ctxT[h][e][d] = f2bfu(ctx[((size_t)(b * NH + h) * NDH + d) * NDH + e]);
  }
  wait_asynccnt0();
  __syncthreads();

  // q softmax over d (axis=-2) * dim_head^-0.5, reading raw [h][d][n] columns
  if (threadIdx.x < NH * 16) {
    int h = threadIdx.x >> 4, n = threadIdx.x & 15;
    float m = -3.0e38f;
#pragma unroll
    for (int d = 0; d < NDH; ++d) m = fmaxf(m, bfu2f(q_raw[h][d][n]));
    float s = 0.0f;
#pragma unroll
    for (int d = 0; d < NDH; ++d) s += __expf(bfu2f(q_raw[h][d][n]) - m);
    float is = 0.17677669529663687f / s;   // scale/sum
#pragma unroll
    for (int d = 0; d < NDH; ++d)
      qs[h][n][d] = f2bfu(__expf(bfu2f(q_raw[h][d][n]) - m) * is);
  }
  __syncthreads();

  // out[e,n] = sum_d ctx[d,e]*qs[d,n]; one 16x16 WMMA tile per wave
  {
    int head = wv >> 1, es = wv & 1;
    v16bf a = frag_a(&ctxT[head][0][0], NDH, es * 16, 0, lane);
    v16bf bq = frag_bt(&qs[head][0][0], NDH, 0, 0, lane);
    v8f oacc = {};
    oacc = wmma_bf16(a, bq, oacc);
    int n = lane & 15;
#pragma unroll
    for (int r = 0; r < 8; ++r) {
      int e = head * NDH + es * 16 + r + 8 * (lane >> 4);
      onb[n][e] = f2bfu(oacc[r]);
    }
  }
  __syncthreads();

  // y = W_out @ out + b_out  (M=256, K=128, N=16); 2 m-subtiles per wave
#pragma unroll
  for (int s = 0; s < 2; ++s) {
    int m0 = (wv + 8 * s) * 16;
    v8f ya = {};
#pragma unroll
    for (int kc = 0; kc < 4; ++kc) {
      v16bf a = frag_a(wout_bf, NHID, m0, kc * 32, lane);
      v16bf bb = frag_bt(&onb[0][0], 132, 0, kc * 32, lane);
      ya = wmma_bf16(a, bb, ya);
    }
    int n = lane & 15;
#pragma unroll
    for (int r = 0; r < 8; ++r) {
      int o = m0 + r + 8 * (lane >> 4);
      yl[o][n] = ya[r] + bout[o];
    }
  }
  __syncthreads();

  // per-pixel RMS norm over 256 channels
  const int n = threadIdx.x & 15, ob = threadIdx.x >> 4;
  {
    float s = 0.0f;
#pragma unroll
    for (int i = 0; i < 16; ++i) { float v = yl[ob * 16 + i][n]; s += v * v; }
    part[ob][n] = s;
  }
  __syncthreads();
  if (threadIdx.x < 16) {
    float ss = 0.0f;
#pragma unroll
    for (int g = 0; g < 16; ++g) ss += part[g][threadIdx.x];
    invn[threadIdx.x] = 1.0f / fmaxf(sqrtf(ss), 1e-12f);
  }
  __syncthreads();
  {
    float iv = invn[n];
#pragma unroll
    for (int i = 0; i < 16; ++i) {
      int o = ob * 16 + i;
      out[((size_t)b * NC + o) * NN + n0 + n] =
          yl[o][n] * iv * (g2[o] + 1.0f) * 16.0f;
    }
  }
}

// ---------------------------------------------------------------------------
// Launcher
// ---------------------------------------------------------------------------
extern "C" void kernel_launch(void* const* d_in, const int* in_sizes, int n_in,
                              void* d_out, int out_size, void* d_ws, size_t ws_size,
                              hipStream_t stream) {
  (void)in_sizes; (void)n_in; (void)out_size; (void)ws_size;
  const float* x     = (const float*)d_in[0];
  const float* g1    = (const float*)d_in[1];
  const float* memkv = (const float*)d_in[2];
  const float* wqkv  = (const float*)d_in[3];
  const float* wout  = (const float*)d_in[4];
  const float* bout  = (const float*)d_in[5];
  const float* g2    = (const float*)d_in[6];
  float* out = (float*)d_out;

  char* ws = (char*)d_ws;
  size_t off = 0;
  unsigned short* wqkv_bf = (unsigned short*)(ws + off); off += (size_t)NO3 * NC * 2;       // 384 KiB
  unsigned short* wout_bf = (unsigned short*)(ws + off); off += (size_t)NC * NHID * 2;      // 64 KiB
  float* rowmax           = (float*)(ws + off);          off += (size_t)NB * NHID * 4;      // 8 KiB
  float* ctxb             = (float*)(ws + off);          off += (size_t)NB * NH * NDH * NDH * 4; // 256 KiB
  off = (off + 255) & ~(size_t)255;
  unsigned short* qw = (unsigned short*)(ws + off); off += (size_t)NB * NHID * NN * 2;      // 64 MiB
  unsigned short* kw = (unsigned short*)(ws + off); off += (size_t)NB * NHID * NN * 2;      // 64 MiB
  unsigned short* vw = (unsigned short*)(ws + off); off += (size_t)NB * NHID * NN * 2;      // 64 MiB

  prep_kernel<<<(NO3 * NC + NC * NHID + 255) / 256, 256, 0, stream>>>(
      wqkv, g1, wout, wqkv_bf, wout_bf);
  qkv_kernel<<<dim3(NN / 32, NB), 256, 0, stream>>>(x, wqkv_bf, qw, kw, vw);
  rowmax_kernel<<<(NB * NHID) / 8, 256, 0, stream>>>(kw, memkv, rowmax);
  context_kernel<<<NB * NH, 128, 0, stream>>>(kw, vw, memkv, rowmax, ctxb);
  attn_out_kernel<<<dim3(NN / 16, NB), 256, 0, stream>>>(
      qw, ctxb, wout_bf, bout, g2, out);
}
